// MambaBlock_15461882265902
// MI455X (gfx1250) — compile-verified, matrix operand order checked by reading, not run on hardware
//
#include <hip/hip_runtime.h>
#include <hip/hip_bf16.h>

// ---------------- problem constants (match reference) ----------------
#define DIMX     1024
#define D_STATE  16
#define D_CONV   4
#define D_INNER  2048
#define DT_RANK  64
#define BATCH    2
#define SEQ      1024
#define BT       (BATCH * SEQ)      // 2048 GEMM rows
#define NCHUNK   16                 // scan chunks
#define TCHUNK   (SEQ / NCHUNK)     // 64 steps per chunk
#define NPROJ    (DT_RANK + 2 * D_STATE)   // 96

typedef __attribute__((ext_vector_type(16))) _Float16 v16h;
typedef __attribute__((ext_vector_type(8)))  _Float16 v8h;
typedef __attribute__((ext_vector_type(8)))  float    v8f;

__device__ __forceinline__ float siluf(float x) {
    return x / (1.0f + __expf(-x));
}
__device__ __forceinline__ float softplusf(float x) {
    return (x > 20.0f) ? x : log1pf(__expf(x));
}

// Load one 16-half operand fragment: two contiguous b128 loads.
__device__ __forceinline__ v16h load_frag16(const _Float16* __restrict__ p) {
    v8h lo = *(const v8h*)(p);
    v8h hi = *(const v8h*)(p + 16);
    return __builtin_shufflevector(lo, hi,
            0, 1, 2, 3, 4, 5, 6, 7, 8, 9, 10, 11, 12, 13, 14, 15);
}

// ---------------------------------------------------------------------
// WMMA GEMM: C(MxN,f32) = A(MxK,f16) @ Bt(NxK,f16 pre-transposed).
// Double-buffered K loop: loads for step k+32 are issued before the
// WMMAs of step k, so VMEM latency hides behind the matrix pipe.
// EPI: 0 = plain store, 1 = softplus(acc + bias[col])
// WH : 1 = also store an f16 copy of C into Ch (same ldc)
// Requires: M % (16*MT) == 0, N % (16*NT) == 0, K % 32 == 0, K >= 32.
// ---------------------------------------------------------------------
template<int MT, int NT, int EPI, int WH>
__global__ void __launch_bounds__(128)
wmma_gemm_kernel(const _Float16* __restrict__ A, int lda,
                 const _Float16* __restrict__ Bt, int ldb,
                 float* __restrict__ C, int ldc,
                 _Float16* __restrict__ Ch,
                 int M, int N, int K,
                 const float* __restrict__ bias)
{
    const int lane = threadIdx.x & 31;
    const int ntg  = N / (16 * NT);
    const int mtg  = M / (16 * MT);
    const int wid  = blockIdx.x * (blockDim.x >> 5) + (threadIdx.x >> 5);
    if (wid >= mtg * ntg) return;
    const int m0   = (wid / ntg) * (16 * MT);
    const int n0   = (wid % ntg) * (16 * NT);
    const int half = lane >> 4;   // 0: lanes 0-15, 1: lanes 16-31
    const int l16  = lane & 15;

    const _Float16* Arow[MT];
    const _Float16* Brow[NT];
#pragma unroll
    for (int mi = 0; mi < MT; ++mi)
        Arow[mi] = A + (long)(m0 + mi * 16 + l16) * lda + half * 8;
#pragma unroll
    for (int ni = 0; ni < NT; ++ni)
        Brow[ni] = Bt + (long)(n0 + ni * 16 + l16) * ldb + half * 8;

    v8f zero = {};
    v8f acc[MT][NT];
#pragma unroll
    for (int mi = 0; mi < MT; ++mi)
#pragma unroll
        for (int ni = 0; ni < NT; ++ni) acc[mi][ni] = zero;

    // prologue: fill buffer 0
    v16h af[2][MT], bf[2][NT];
#pragma unroll
    for (int mi = 0; mi < MT; ++mi) af[0][mi] = load_frag16(Arow[mi]);
#pragma unroll
    for (int ni = 0; ni < NT; ++ni) bf[0][ni] = load_frag16(Brow[ni]);

    int k0 = 0;
    for (; k0 < K - 32; k0 += 32) {
        const int cur = (k0 >> 5) & 1;
        const int nxt = cur ^ 1;
        // issue next-step loads before consuming current fragments
#pragma unroll
        for (int mi = 0; mi < MT; ++mi) af[nxt][mi] = load_frag16(Arow[mi] + k0 + 32);
#pragma unroll
        for (int ni = 0; ni < NT; ++ni) bf[nxt][ni] = load_frag16(Brow[ni] + k0 + 32);
#pragma unroll
        for (int ni = 0; ni < NT; ++ni)
#pragma unroll
            for (int mi = 0; mi < MT; ++mi)
                acc[mi][ni] = __builtin_amdgcn_wmma_f32_16x16x32_f16(
                    false, af[cur][mi], false, bf[cur][ni], (short)0,
                    acc[mi][ni], false, false);
    }
    {   // epilogue K-step
        const int cur = (k0 >> 5) & 1;
#pragma unroll
        for (int ni = 0; ni < NT; ++ni)
#pragma unroll
            for (int mi = 0; mi < MT; ++mi)
                acc[mi][ni] = __builtin_amdgcn_wmma_f32_16x16x32_f16(
                    false, af[cur][mi], false, bf[cur][ni], (short)0,
                    acc[mi][ni], false, false);
    }

    // Store: C/D layout — VGPR r holds row m0 + r + half*8, col = n0 + l16.
#pragma unroll
    for (int mi = 0; mi < MT; ++mi) {
#pragma unroll
        for (int ni = 0; ni < NT; ++ni) {
            const int col = n0 + ni * 16 + l16;
            const long rbase = (long)(m0 + mi * 16 + half * 8);
            float* Cp = C + rbase * ldc + col;
            if (EPI == 1) {
                const float bvv = bias[col];
#pragma unroll
                for (int r = 0; r < 8; ++r) {
                    const float v = softplusf(acc[mi][ni][r] + bvv);
                    Cp[(long)r * ldc] = v;
                    if (WH) Ch[(rbase + r) * ldc + col] = (_Float16)v;
                }
            } else {
#pragma unroll
                for (int r = 0; r < 8; ++r) {
                    const float v = acc[mi][ni][r];
                    Cp[(long)r * ldc] = v;
                    if (WH) Ch[(rbase + r) * ldc + col] = (_Float16)v;
                }
            }
        }
    }
}

// ---------------------------------------------------------------------
// Prep: W (KxN, f32) -> Wt (NxK, f16), LDS-tiled 32x32 transpose.
// K, N multiples of 32 (holds for all four weights).
// ---------------------------------------------------------------------
__global__ void __launch_bounds__(256)
transpose_f32_to_f16(const float* __restrict__ src, _Float16* __restrict__ dst,
                     int K, int N)
{
    __shared__ float tile[32][33];
    const int tx = threadIdx.x & 31;
    const int ty = threadIdx.x >> 5;            // 0..7
    const int n0 = blockIdx.x * 32;
    const int k0 = blockIdx.y * 32;
#pragma unroll
    for (int r = ty; r < 32; r += 8)
        tile[r][tx] = src[(long)(k0 + r) * N + n0 + tx];
    __syncthreads();
#pragma unroll
    for (int r = ty; r < 32; r += 8)
        dst[(long)(n0 + r) * K + k0 + tx] = (_Float16)tile[tx][r];
}

__global__ void __launch_bounds__(256)
f32_to_f16_kernel(const float* __restrict__ src, _Float16* __restrict__ dst, int n)
{
    const int i = blockIdx.x * 256 + threadIdx.x;
    if (i < n) dst[i] = (_Float16)src[i];
}

// ---------------------------------------------------------------------
// Depthwise causal conv (d_conv=4, zero initial state) + SiLU.
// Writes xc (f32, scan input) and xch (f16, GEMM operand).
// ---------------------------------------------------------------------
__global__ void __launch_bounds__(256)
conv_silu_kernel(const float* __restrict__ xz,
                 const float* __restrict__ conv_w,
                 const float* __restrict__ conv_b,
                 float* __restrict__ xc,
                 _Float16* __restrict__ xch)
{
    const int idx = blockIdx.x * 256 + threadIdx.x;
    if (idx >= BT * D_INNER) return;
    const int d   = idx & (D_INNER - 1);
    const int row = idx >> 11;          // b*SEQ + t
    const int t   = row & (SEQ - 1);
    const float w0 = conv_w[d * 4 + 0], w1 = conv_w[d * 4 + 1];
    const float w2 = conv_w[d * 4 + 2], w3 = conv_w[d * 4 + 3];
    const float* base = xz + (long)row * (2 * D_INNER) + d;
    float acc = conv_b[d] + base[0] * w3;                   // k=3 -> t
    if (t >= 1) acc += base[-(long)(2 * D_INNER)] * w2;     // t-1
    if (t >= 2) acc += base[-(long)(4 * D_INNER)] * w1;     // t-2
    if (t >= 3) acc += base[-(long)(6 * D_INNER)] * w0;     // t-3
    const float v = siluf(acc);
    xc[idx]  = v;
    xch[idx] = (_Float16)v;
}

// new_conv_state = swapaxes(x_in[:, T-3:, :], 1, 2)  -> (B, Di, 3)
__global__ void __launch_bounds__(256)
conv_state_kernel(const float* __restrict__ xz, float* __restrict__ cs)
{
    const int idx = blockIdx.x * 256 + threadIdx.x;
    if (idx >= BATCH * D_INNER * (D_CONV - 1)) return;
    const int j = idx % 3;
    const int d = (idx / 3) % D_INNER;
    const int b = idx / (3 * D_INNER);
    cs[idx] = xz[(long)(b * SEQ + SEQ - 3 + j) * (2 * D_INNER) + d];
}

// ---------------------------------------------------------------------
// Chunk-parallel selective scan.
// h_t = exp(dt_t * a_n) * h_{t-1} + (dt_t * B_t[n]) * x_t
// Chunk transition product over n is exp(a_n * sum(dt)) — scalar sum.
// P/HL/HI layout: [(c*BATCH + b)*16 + n]*D_INNER + d   (coalesced in d)
// ---------------------------------------------------------------------
__device__ __forceinline__ long pidx(int c, int b, int n, int d) {
    return ((long)((c * BATCH + b) * 16 + n)) * D_INNER + d;
}

// Phase 1: per-chunk local scan with h0 = 0; emit local state + transition.
__global__ void __launch_bounds__(256)
scan_phase1(const float* __restrict__ dt, const float* __restrict__ xc,
            const float* __restrict__ proj, const float* __restrict__ A_log,
            float* __restrict__ P, float* __restrict__ HL)
{
    __shared__ float sB[TCHUNK * 16];
    const int blk  = blockIdx.x;                // (c, b, dblk)
    const int dblk = blk & 7;
    const int b    = (blk >> 3) & (BATCH - 1);
    const int c    = blk >> 4;
    const int d    = dblk * 256 + threadIdx.x;
    const int t0   = c * TCHUNK;

    for (int i = threadIdx.x; i < TCHUNK * 16; i += 256) {
        const int tt = i >> 4, j = i & 15;
        sB[i] = proj[(long)(b * SEQ + t0 + tt) * NPROJ + DT_RANK + j];
    }
    __syncthreads();

    float a[16], h[16];
#pragma unroll
    for (int n = 0; n < 16; ++n) { a[n] = -__expf(A_log[d * 16 + n]); h[n] = 0.0f; }
    float dtsum = 0.0f;

    for (int tl = 0; tl < TCHUNK; ++tl) {
        const long row = (long)(b * SEQ + t0 + tl);
        const float dtv = dt[row * D_INNER + d];
        const float xv  = xc[row * D_INNER + d];
        dtsum += dtv;
#pragma unroll
        for (int n = 0; n < 16; ++n) {
            const float dA = __expf(dtv * a[n]);
            h[n] = dA * h[n] + (dtv * sB[tl * 16 + n]) * xv;
        }
    }
#pragma unroll
    for (int n = 0; n < 16; ++n) {
        P[pidx(c, b, n, d)]  = __expf(a[n] * dtsum);
        HL[pidx(c, b, n, d)] = h[n];
    }
}

// Phase 2: sequential combine over the 16 chunks; emits per-chunk initial
// states and the final hidden state output.
__global__ void __launch_bounds__(256)
scan_phase2(const float* __restrict__ P, const float* __restrict__ HL,
            float* __restrict__ HI, float* __restrict__ hfinal)
{
    const int b = blockIdx.x >> 3;
    const int d = ((blockIdx.x & 7) << 8) + threadIdx.x;
    float h[16];
#pragma unroll
    for (int n = 0; n < 16; ++n) h[n] = 0.0f;
    for (int c = 0; c < NCHUNK; ++c) {
#pragma unroll
        for (int n = 0; n < 16; ++n) {
            const long i = pidx(c, b, n, d);
            HI[i] = h[n];
            h[n] = P[i] * h[n] + HL[i];
        }
    }
#pragma unroll
    for (int n = 0; n < 16; ++n)
        hfinal[(long)(b * D_INNER + d) * 16 + n] = h[n];
}

// Phase 3: replay each chunk from its true initial state, emit
// ygh = (y + D*x) * silu(z) in f16 (operand of the output GEMM).
__global__ void __launch_bounds__(256)
scan_phase3(const float* __restrict__ dt, const float* __restrict__ xc,
            const float* __restrict__ xz,   // z = cols [2048, 4096)
            const float* __restrict__ proj, const float* __restrict__ A_log,
            const float* __restrict__ Dparam, const float* __restrict__ HI,
            _Float16* __restrict__ ygh)
{
    __shared__ float sB[TCHUNK * 16];
    __shared__ float sC[TCHUNK * 16];
    const int blk  = blockIdx.x;
    const int dblk = blk & 7;
    const int b    = (blk >> 3) & (BATCH - 1);
    const int c    = blk >> 4;
    const int d    = dblk * 256 + threadIdx.x;
    const int t0   = c * TCHUNK;

    for (int i = threadIdx.x; i < TCHUNK * 32; i += 256) {
        const int tt = i >> 5, j = i & 31;
        const float v = proj[(long)(b * SEQ + t0 + tt) * NPROJ + DT_RANK + j];
        if (j < 16) sB[tt * 16 + j] = v;
        else        sC[tt * 16 + (j - 16)] = v;
    }
    __syncthreads();

    float a[16], h[16];
#pragma unroll
    for (int n = 0; n < 16; ++n) {
        a[n] = -__expf(A_log[d * 16 + n]);
        h[n] = HI[pidx(c, b, n, d)];
    }
    const float Dv = Dparam[d];

    for (int tl = 0; tl < TCHUNK; ++tl) {
        const long row = (long)(b * SEQ + t0 + tl);
        const float dtv = dt[row * D_INNER + d];
        const float xv  = xc[row * D_INNER + d];
        const float zv  = xz[row * (2 * D_INNER) + D_INNER + d];
        float y = 0.0f;
#pragma unroll
        for (int n = 0; n < 16; ++n) {
            const float dA = __expf(dtv * a[n]);
            h[n] = dA * h[n] + (dtv * sB[tl * 16 + n]) * xv;
            y += h[n] * sC[tl * 16 + n];
        }
        y += Dv * xv;
        ygh[row * D_INNER + d] = (_Float16)(y * siluf(zv));
    }
}

// ---------------------------------------------------------------------
extern "C" void kernel_launch(void* const* d_in, const int* in_sizes, int n_in,
                              void* d_out, int out_size, void* d_ws, size_t ws_size,
                              hipStream_t stream)
{
    (void)in_sizes; (void)n_in; (void)out_size; (void)ws_size;
    const float* x      = (const float*)d_in[0];
    const float* W_in   = (const float*)d_in[1];
    const float* conv_w = (const float*)d_in[2];
    const float* conv_b = (const float*)d_in[3];
    const float* W_x    = (const float*)d_in[4];
    const float* W_dt   = (const float*)d_in[5];
    const float* b_dt   = (const float*)d_in[6];
    const float* A_log  = (const float*)d_in[7];
    const float* D_par  = (const float*)d_in[8];
    const float* W_out  = (const float*)d_in[9];

    float* out    = (float*)d_out;                                  // (B,T,DIM)
    float* hfinal = out + (long)BATCH * SEQ * DIMX;                 // (B,Di,N)
    float* cstate = hfinal + (long)BATCH * D_INNER * D_STATE;       // (B,Di,3)

    // ---- workspace layout: f32 region then f16 region ----
    float* ws   = (float*)d_ws;
    float* xz   = ws;                                        // BT x 4096
    float* xc   = xz + (long)BT * 2 * D_INNER;               // BT x 2048
    float* proj = xc + (long)BT * D_INNER;                   // BT x 96
    float* dt   = proj + (long)BT * NPROJ;                   // BT x 2048
    float* P    = dt + (long)BT * D_INNER;                   // NC*B*16*Di
    float* HL   = P + (long)NCHUNK * BATCH * 16 * D_INNER;
    float* HI   = HL + (long)NCHUNK * BATCH * 16 * D_INNER;

    _Float16* xh    = (_Float16*)(HI + (long)NCHUNK * BATCH * 16 * D_INNER);
    _Float16* xch   = xh + (long)BT * DIMX;                  // BT x 2048
    _Float16* projh = xch + (long)BT * D_INNER;              // BT x 96
    _Float16* ygh   = projh + (long)BT * NPROJ;              // BT x 2048
    _Float16* Wt_in = ygh + (long)BT * D_INNER;              // 4096 x 1024
    _Float16* Wt_x  = Wt_in + (long)DIMX * 2 * D_INNER;      // 96 x 2048
    _Float16* Wt_dt = Wt_x + (long)D_INNER * NPROJ;          // 2048 x 64
    _Float16* Wt_out= Wt_dt + (long)DT_RANK * D_INNER;       // 1024 x 2048

    auto gemm_blocks = [](int M, int N, int MT, int NT) {
        const int groups = (M / (16 * MT)) * (N / (16 * NT));
        return (groups + 3) / 4;   // 4 waves per 128-thread block
    };

    // 0) prep: convert x, transpose+convert all weights to f16
    f32_to_f16_kernel<<<(BT * DIMX + 255) / 256, 256, 0, stream>>>(x, xh, BT * DIMX);
    transpose_f32_to_f16<<<dim3(2 * D_INNER / 32, DIMX / 32), 256, 0, stream>>>(
        W_in, Wt_in, DIMX, 2 * D_INNER);
    transpose_f32_to_f16<<<dim3(NPROJ / 32, D_INNER / 32), 256, 0, stream>>>(
        W_x, Wt_x, D_INNER, NPROJ);
    transpose_f32_to_f16<<<dim3(D_INNER / 32, DT_RANK / 32), 256, 0, stream>>>(
        W_dt, Wt_dt, DT_RANK, D_INNER);
    transpose_f32_to_f16<<<dim3(DIMX / 32, D_INNER / 32), 256, 0, stream>>>(
        W_out, Wt_out, D_INNER, DIMX);

    // 1) xz = x @ W_in            (2048 x 1024 x 4096)
    wmma_gemm_kernel<2, 4, 0, 0><<<gemm_blocks(BT, 2 * D_INNER, 2, 4), 128, 0, stream>>>(
        xh, DIMX, Wt_in, DIMX, xz, 2 * D_INNER, nullptr, BT, 2 * D_INNER, DIMX, nullptr);

    // 2) depthwise conv + SiLU (f32 + f16 outputs), and new_conv_state
    conv_silu_kernel<<<(BT * D_INNER + 255) / 256, 256, 0, stream>>>(
        xz, conv_w, conv_b, xc, xch);
    conv_state_kernel<<<(BATCH * D_INNER * 3 + 255) / 256, 256, 0, stream>>>(xz, cstate);

    // 3) proj = xc @ W_x          (2048 x 2048 x 96), also emit f16 copy
    wmma_gemm_kernel<2, 2, 0, 1><<<gemm_blocks(BT, NPROJ, 2, 2), 128, 0, stream>>>(
        xch, D_INNER, Wt_x, D_INNER, proj, NPROJ, projh, BT, NPROJ, D_INNER, nullptr);

    // 4) dt = softplus(proj[:, :64] @ W_dt + b_dt)   (2048 x 64 x 2048)
    wmma_gemm_kernel<2, 4, 1, 0><<<gemm_blocks(BT, D_INNER, 2, 4), 128, 0, stream>>>(
        projh, NPROJ, Wt_dt, DT_RANK, dt, D_INNER, nullptr, BT, D_INNER, DT_RANK, b_dt);

    // 5) chunk-parallel scan -> ygh (f16), h_final to output
    scan_phase1<<<NCHUNK * BATCH * (D_INNER / 256), 256, 0, stream>>>(
        dt, xc, proj, A_log, P, HL);
    scan_phase2<<<BATCH * (D_INNER / 256), 256, 0, stream>>>(P, HL, HI, hfinal);
    scan_phase3<<<NCHUNK * BATCH * (D_INNER / 256), 256, 0, stream>>>(
        dt, xc, xz, proj, A_log, D_par, HI, ygh);

    // 6) out = yg @ W_out         (2048 x 2048 x 1024)
    wmma_gemm_kernel<2, 4, 0, 0><<<gemm_blocks(BT, DIMX, 2, 4), 128, 0, stream>>>(
        ygh, D_INNER, Wt_out, D_INNER, out, DIMX, nullptr, BT, DIMX, D_INNER, nullptr);
}
